// Attention_71339406787178
// MI455X (gfx1250) — matrix operand order repaired and checked by reading, not compiled, not run on hardware
//
#include <hip/hip_runtime.h>
#include <stdint.h>

// Problem constants (match the reference)
constexpr int kDim     = 1024;   // model dim
constexpr int kHeads   = 16;
constexpr int kDimHead = 64;
constexpr int kInner   = kHeads * kDimHead;   // 1024
constexpr int kBatch   = 2;
constexpr int kSeq     = 2048;
constexpr int kRows    = kBatch * kSeq;       // 4096 flattened rows
constexpr float kScale = 0.125f;              // 64^-0.5

typedef __attribute__((ext_vector_type(16))) _Float16 v16h;
typedef __attribute__((ext_vector_type(8)))  float    v8f;
typedef int vi4 __attribute__((vector_size(4 * sizeof(int))));  // 16B chunk

// ---- Optional CDNA5 async global->LDS path (guarded; sync fallback) ----
#if __has_builtin(__builtin_amdgcn_global_load_async_to_lds_b128) && \
    __has_builtin(__builtin_amdgcn_s_wait_asynccnt)
#define USE_ASYNC_LDS 1
#else
#define USE_ASYNC_LDS 0
#endif

#if USE_ASYNC_LDS
__device__ __forceinline__ void async_b128(const void* g, void* l) {
  // global_load_async_to_lds_b128: per-lane 16B global -> LDS, ASYNCcnt-tracked
  __builtin_amdgcn_global_load_async_to_lds_b128(
      (__attribute__((address_space(1))) vi4*)(uintptr_t)g,
      (__attribute__((address_space(3))) vi4*)(uintptr_t)l, 0, 0);
}
#endif

// ---- Optional CDNA5 Tensor Data Mover path (guarded) ----
#if __has_builtin(__builtin_amdgcn_tensor_load_to_lds) && \
    __has_builtin(__builtin_amdgcn_s_wait_tensorcnt)
#define USE_TDM 1
#else
#define USE_TDM 0
#endif

#if USE_TDM
typedef unsigned int v4u __attribute__((ext_vector_type(4)));
typedef int          v8i __attribute__((ext_vector_type(8)));
typedef int          v4i __attribute__((ext_vector_type(4)));

// One-shot 2D tile DMA: rows x cols f16 elements, global row stride
// row_stride_elems, into LDS with 16B padding appended per 128B row
// (pad_interval=32 DWORDs, pad_amount=4 DWORDs -> LDS row pitch 144B).
// Issue from a single wave; tracked by TENSORcnt.  (ISA §7.2/§8)
__device__ __forceinline__ void tdm_load_2d_f16(const void* gaddr, void* lds,
                                                int rows, int cols_elems,
                                                int row_stride_elems) {
  const uint64_t ga = (uint64_t)(uintptr_t)gaddr;
  const uint32_t la = (uint32_t)(uintptr_t)lds;
  v4u g0;
  g0[0] = 1u;                                   // count=1 (valid user D#)
  g0[1] = la;                                   // lds_addr
  g0[2] = (uint32_t)ga;                         // global_addr[31:0]
  g0[3] = (uint32_t)((ga >> 32) & 0x01FFFFFFu)  // global_addr[56:32]
          | 0x80000000u;                        // type=2 ("image")
  v8i g1;
  g1[0] = (int)((1u << 16)      // data_size = 2 bytes
                | (1u << 20)    // pad_enable
                | (4u << 22)    // pad_interval: 32 DWORDs (=128B row)
                | (3u << 25));  // pad_amount: 4 DWORDs (=16B)
  g1[1] = (int)(((uint32_t)cols_elems & 0xFFFFu) << 16);      // tensor_dim0 lo
  g1[2] = (int)((((uint32_t)cols_elems >> 16) & 0xFFFFu) |
                (((uint32_t)rows & 0xFFFFu) << 16));          // dim0 hi/dim1 lo
  g1[3] = (int)((((uint32_t)rows >> 16) & 0xFFFFu) |
                ((uint32_t)cols_elems << 16));                // dim1 hi/tile_dim0
  g1[4] = rows;                                               // tile_dim1
  g1[5] = row_stride_elems;                                   // dim0_stride lo32
  g1[6] = 0;
  g1[7] = 0;
  v4i z4 = {0, 0, 0, 0};
  v8i z8 = {0, 0, 0, 0, 0, 0, 0, 0};
  // 6-arg toolchain form: (g0, g1, g2, g3, extra, cpol)
  __builtin_amdgcn_tensor_load_to_lds(g0, g1, z4, z4, z8, 0);
}
#endif

// ---- WMMA fragment index maps (CDNA5 ISA 7.12.2 / 7.12.4, wave32) ----
// A-matrix 16-bit 16x32: lane L holds row L%16; element j -> K:
//   lanes 0-15:  K in {0..7, 16..23};  lanes 16-31: K in {8..15, 24..31}
__device__ __forceinline__ int amap(int j, int half) {
  return ((j < 8) ? 0 : 16) + half * 8 + (j & 7);
}
// B-matrix 16-bit 32x16: lane L holds column L%16; element j -> K:
//   lanes 0-15:  K in {0..7, 16..23};  lanes 16-31: K in {8..15, 24..31}
__device__ __forceinline__ int bmap(int j, int half) {
  return ((j < 8) ? 0 : 16) + half * 8 + (j & 7);
}
// C/D 16x16 f32: VGPR r, lane L -> (M = r + 8*(L/16), N = L%16)

__device__ __forceinline__ v8f wmma16x16x32(v16h a, v16h b, v8f c) {
  // (neg_a, A, neg_b, B, c_mod, C, reuse_a, reuse_b)
  return __builtin_amdgcn_wmma_f32_16x16x32_f16(false, a, false, b, (short)0, c,
                                                false, false);
}

// ============================================================================
// f32 -> f16 conversion (one-time prep so GEMM tiles can be DMA'd to LDS)
// ============================================================================
__global__ __launch_bounds__(256)
void f32_to_f16_kernel(const float* __restrict__ in, _Float16* __restrict__ out,
                       int n) {
  const int i = blockIdx.x * 256 + threadIdx.x;
  if (i < n) out[i] = (_Float16)in[i];
}

// ============================================================================
// Tiled WMMA GEMM: C[M,Nc] = A[M,K] * W[K,Nc], A/W f16, C f16 or f32.
// Block = 128 threads (4 waves); block tile = 64x64; K-step = 32.
// A tile: per-lane async b128 DMA.  W tile: single TDM descriptor DMA.
// ============================================================================
template <typename OT>
__global__ __launch_bounds__(128)
void gemm_wmma_kernel(const _Float16* __restrict__ A,
                      const _Float16* __restrict__ W, OT* __restrict__ C, int M,
                      int K, int Nc) {
  // Row strides padded to a 16-byte multiple (async/TDM chunks stay aligned)
  constexpr int ASTR = 40;  // 32 + 8
  constexpr int BSTR = 72;  // 64 + 8  (matches TDM pad: 128B data + 16B pad)
  __shared__ __align__(16) _Float16 As[64 * ASTR];
  __shared__ __align__(16) _Float16 Bs[32 * BSTR];

  const int tid  = threadIdx.x;
  const int wave = tid >> 5;
  const int lane = tid & 31;
  const int half = lane >> 4;
  const int lrow = lane & 15;
  const int m0 = blockIdx.y * 64;
  const int n0 = blockIdx.x * 64;

  v8f acc0 = {}, acc1 = {}, acc2 = {}, acc3 = {};

  for (int k0 = 0; k0 < K; k0 += 32) {
    const _Float16* abase = A + (size_t)m0 * K + k0;   // 64 x 32 tile
    const _Float16* wbase = W + (size_t)k0 * Nc + n0;  // 32 x 64 tile

    // ---- W tile: TDM if available, else async, else sync ----
#if USE_TDM
    if (wave == 0) tdm_load_2d_f16(wbase, Bs, 32, 64, Nc);
#elif USE_ASYNC_LDS
#pragma unroll
    for (int c = tid; c < 256; c += 128) {
      const int row = c >> 3, col = (c & 7) * 8;
      async_b128(wbase + (size_t)row * Nc + col, &Bs[row * BSTR + col]);
    }
#else
    {
      const int row = tid >> 2, seg = (tid & 3) * 16;
      const _Float16* wp = wbase + (size_t)row * Nc + seg;
#pragma unroll
      for (int e = 0; e < 16; ++e) Bs[row * BSTR + seg + e] = wp[e];
    }
#endif

    // ---- A tile: async per-lane b128 DMA (2 chunks per thread) ----
#if USE_ASYNC_LDS
#pragma unroll
    for (int c = tid; c < 256; c += 128) {
      const int row = c >> 2, col = (c & 3) * 8;
      async_b128(abase + (size_t)row * K + col, &As[row * ASTR + col]);
    }
#else
    {
      const int row = tid >> 1, seg = (tid & 1) * 16;
      const _Float16* ap = abase + (size_t)row * K + seg;
#pragma unroll
      for (int e = 0; e < 16; ++e) As[row * ASTR + seg + e] = ap[e];
    }
#endif

    // Prefetch next k-tile (lowers to global_prefetch_b8)
    if (k0 + 32 < K) {
      __builtin_prefetch(abase + 32 + (size_t)(tid >> 1) * K, 0, 1);
      __builtin_prefetch(wbase + (size_t)(32 + (tid >> 2)) * Nc, 0, 1);
    }

#if USE_ASYNC_LDS
    __builtin_amdgcn_s_wait_asynccnt(0);
#endif
#if USE_TDM
    if (wave == 0) __builtin_amdgcn_s_wait_tensorcnt(0);
#endif
    __syncthreads();

    // Build fragments and issue 4 WMMAs (one 16-row strip x 64 cols per wave)
    v16h af;
#pragma unroll
    for (int j = 0; j < 16; ++j)
      af[j] = As[(wave * 16 + lrow) * ASTR + amap(j, half)];

    v16h bf0, bf1, bf2, bf3;
#pragma unroll
    for (int j = 0; j < 16; ++j) {
      const int kk = bmap(j, half);
      bf0[j] = Bs[kk * BSTR + lrow];
      bf1[j] = Bs[kk * BSTR + 16 + lrow];
      bf2[j] = Bs[kk * BSTR + 32 + lrow];
      bf3[j] = Bs[kk * BSTR + 48 + lrow];
    }
    acc0 = wmma16x16x32(af, bf0, acc0);
    acc1 = wmma16x16x32(af, bf1, acc1);
    acc2 = wmma16x16x32(af, bf2, acc2);
    acc3 = wmma16x16x32(af, bf3, acc3);
    __syncthreads();
  }

  // Epilogue: C/D layout -> global
#pragma unroll
  for (int r = 0; r < 8; ++r) {
    const size_t row = (size_t)(m0 + wave * 16 + r + 8 * half);
    OT* cp = C + row * Nc + n0;
    cp[lrow]      = (OT)acc0[r];
    cp[16 + lrow] = (OT)acc1[r];
    cp[32 + lrow] = (OT)acc2[r];
    cp[48 + lrow] = (OT)acc3[r];
  }
}

// ============================================================================
// Flash attention (causal), one wave per (b, h, 16 query rows).
// Per 32-key tile: S(16x32) via 4 WMMAs, online softmax via LDS row stats,
// P re-laid out through LDS as one A-fragment, O(16x64) += P*V via 4 WMMAs.
// ============================================================================
__global__ __launch_bounds__(32)
void flash_attn_kernel(const _Float16* __restrict__ Q,
                       const _Float16* __restrict__ Kb,
                       const _Float16* __restrict__ V,
                       _Float16* __restrict__ O) {
  const int qt = blockIdx.x, h = blockIdx.y, b = blockIdx.z;
  const int q0 = qt * 16;
  const int lane = threadIdx.x;
  const int half = lane >> 4;
  const int lrow = lane & 15;
  const size_t rs = kInner;  // row stride of Q/K/V/O buffers

  __shared__ float    s_tile[16][32];
  __shared__ _Float16 p_tile[16][32];
  __shared__ float m_s[16], l_s[16], a_s[16];

  const _Float16* qbase = Q + ((size_t)(b * kSeq + q0)) * kInner + h * kDimHead;

  // Q A-fragments for d-chunks [0,32) and [32,64)
  v16h aq0, aq1;
#pragma unroll
  for (int j = 0; j < 16; ++j) {
    aq0[j] = qbase[(size_t)lrow * rs + amap(j, half)];
    aq1[j] = qbase[(size_t)lrow * rs + 32 + amap(j, half)];
  }

  v8f o0 = {}, o1 = {}, o2 = {}, o3 = {};
  if (lane < 16) { m_s[lane] = -1e30f; l_s[lane] = 0.f; }
  __syncthreads();

  // Causal: only key tiles with k0 <= q0+15 contribute (k0 <= q0 in practice)
  for (int k0 = 0; k0 <= q0 + 15; k0 += 32) {
    const _Float16* kbase =
        Kb + ((size_t)(b * kSeq + k0)) * kInner + h * kDimHead;

    // ---- S = Q * K^T  (two 16x16 C tiles, contract d=64 in two chunks) ----
    v8f s0 = {}, s1 = {};
    {
      v16h b0, b1;
#pragma unroll
      for (int j = 0; j < 16; ++j) {
        const int kk = bmap(j, half);
        b0[j] = kbase[(size_t)lrow * rs + kk];          // keys k0+0..15
        b1[j] = kbase[(size_t)(lrow + 16) * rs + kk];   // keys k0+16..31
      }
      s0 = wmma16x16x32(aq0, b0, s0);
      s1 = wmma16x16x32(aq0, b1, s1);
#pragma unroll
      for (int j = 0; j < 16; ++j) {
        const int kk = 32 + bmap(j, half);
        b0[j] = kbase[(size_t)lrow * rs + kk];
        b1[j] = kbase[(size_t)(lrow + 16) * rs + kk];
      }
      s0 = wmma16x16x32(aq1, b0, s0);
      s1 = wmma16x16x32(aq1, b1, s1);
    }

    // Spill S (scaled) into LDS in row-major form
#pragma unroll
    for (int r = 0; r < 8; ++r) {
      s_tile[r + 8 * half][lrow]      = s0[r] * kScale;
      s_tile[r + 8 * half][16 + lrow] = s1[r] * kScale;
    }
    __syncthreads();

    // ---- Online softmax; one lane per query row; causal mask here ----
    if (lane < 16) {
      const int rg = q0 + lane;          // global query row
      int cmax = rg - k0;                // last unmasked column in this tile
      if (cmax > 31) cmax = 31;
      const float mo = m_s[lane];
      float mn = mo;
      for (int c = 0; c <= cmax; ++c) mn = fmaxf(mn, s_tile[lane][c]);
      const float alpha = __expf(mo - mn);
      float ln = alpha * l_s[lane];
#pragma unroll
      for (int c = 0; c < 32; ++c) {
        const float p = (c <= cmax) ? __expf(s_tile[lane][c] - mn) : 0.f;
        p_tile[lane][c] = (_Float16)p;
        ln += p;
      }
      m_s[lane] = mn; l_s[lane] = ln; a_s[lane] = alpha;
    }
    __syncthreads();

    // Rescale running O by alpha(row)
#pragma unroll
    for (int r = 0; r < 8; ++r) {
      const float al = a_s[r + 8 * half];
      o0[r] *= al; o1[r] *= al; o2[r] *= al; o3[r] *= al;
    }

    // ---- O += P * V  (P is exactly one 16x32 A fragment) ----
    v16h pf;
#pragma unroll
    for (int j = 0; j < 16; ++j) pf[j] = p_tile[lrow][amap(j, half)];

    const _Float16* vbase =
        V + ((size_t)(b * kSeq + k0)) * kInner + h * kDimHead;
    v16h vf0, vf1, vf2, vf3;
#pragma unroll
    for (int j = 0; j < 16; ++j) {
      const int kk = bmap(j, half);       // key index within the tile
      const _Float16* vr = vbase + (size_t)kk * rs;
      vf0[j] = vr[lrow];
      vf1[j] = vr[16 + lrow];
      vf2[j] = vr[32 + lrow];
      vf3[j] = vr[48 + lrow];
    }
    o0 = wmma16x16x32(pf, vf0, o0);
    o1 = wmma16x16x32(pf, vf1, o1);
    o2 = wmma16x16x32(pf, vf2, o2);
    o3 = wmma16x16x32(pf, vf3, o3);
    __syncthreads();
  }

  // Finalize: divide by row sums, write f16 attention output
  _Float16* obase = O + ((size_t)(b * kSeq + q0)) * kInner + h * kDimHead;
#pragma unroll
  for (int r = 0; r < 8; ++r) {
    const float inv = 1.f / l_s[r + 8 * half];
    _Float16* op = obase + (size_t)(r + 8 * half) * rs;
    op[lrow]      = (_Float16)(o0[r] * inv);
    op[16 + lrow] = (_Float16)(o1[r] * inv);
    op[32 + lrow] = (_Float16)(o2[r] * inv);
    op[48 + lrow] = (_Float16)(o3[r] * inv);
  }
}

// ============================================================================
// Launch: f16 conversions -> 3 projection GEMMs -> flash attention -> out proj
// ws layout (f16): xh(4M) | Wq..Wo (4x1M) | Q | K | V | attn (4x4M) = 48 MB
// ============================================================================
extern "C" void kernel_launch(void* const* d_in, const int* in_sizes, int n_in,
                              void* d_out, int out_size, void* d_ws,
                              size_t ws_size, hipStream_t stream) {
  (void)in_sizes; (void)n_in; (void)out_size; (void)ws_size;
  const float* x  = (const float*)d_in[0];
  const float* Wq = (const float*)d_in[1];
  const float* Wk = (const float*)d_in[2];
  const float* Wv = (const float*)d_in[3];
  const float* Wo = (const float*)d_in[4];

  const size_t xel = (size_t)kRows * kDim;    // 4M
  const size_t wel = (size_t)kDim * kInner;   // 1M
  _Float16* xh  = (_Float16*)d_ws;
  _Float16* Wqh = xh + xel;
  _Float16* Wkh = Wqh + wel;
  _Float16* Wvh = Wkh + wel;
  _Float16* Woh = Wvh + wel;
  _Float16* Qh  = Woh + wel;
  _Float16* Kh  = Qh + xel;
  _Float16* Vh  = Kh + xel;
  _Float16* Ah  = Vh + xel;

  f32_to_f16_kernel<<<(int)(xel / 256), 256, 0, stream>>>(x, xh, (int)xel);
  f32_to_f16_kernel<<<(int)(wel / 256), 256, 0, stream>>>(Wq, Wqh, (int)wel);
  f32_to_f16_kernel<<<(int)(wel / 256), 256, 0, stream>>>(Wk, Wkh, (int)wel);
  f32_to_f16_kernel<<<(int)(wel / 256), 256, 0, stream>>>(Wv, Wvh, (int)wel);
  f32_to_f16_kernel<<<(int)(wel / 256), 256, 0, stream>>>(Wo, Woh, (int)wel);

  dim3 ggrid(kInner / 64, kRows / 64);  // (16, 64)
  gemm_wmma_kernel<_Float16>
      <<<ggrid, 128, 0, stream>>>(xh, Wqh, Qh, kRows, kDim, kInner);
  gemm_wmma_kernel<_Float16>
      <<<ggrid, 128, 0, stream>>>(xh, Wkh, Kh, kRows, kDim, kInner);
  gemm_wmma_kernel<_Float16>
      <<<ggrid, 128, 0, stream>>>(xh, Wvh, Vh, kRows, kDim, kInner);

  flash_attn_kernel<<<dim3(kSeq / 16, kHeads, kBatch), 32, 0, stream>>>(
      Qh, Kh, Vh, Ah);

  gemm_wmma_kernel<float>
      <<<ggrid, 128, 0, stream>>>(Ah, Woh, (float*)d_out, kRows, kInner, kDim);
}
